// SLSTM_91147795956280
// MI455X (gfx1250) — matrix operand order repaired and checked by reading, not compiled, not run on hardware
//
#include <hip/hip_runtime.h>
#include <cstdint>

// ---------------- problem constants ----------------
#define S_LEN 1024
#define BATCH 16
#define DIM   1024
#define INF_  1024          // input feature dim
#define FOURD 4096
#define NBLK  32            // persistent blocks; each owns 128 N-rows of W1
#define TPB   256           // 8 waves of 32
#define EPS_F 1e-6f

// LDS layout for the recurrence kernel (row padding 1024->1032 elements:
// row stride 2064 B = 516 dwords -> consecutive rows land 4 banks apart,
// so 16 lanes reading 16 different rows hit 16 distinct 4-bank groups).
#define W1ROW   1032
#define H0ROW   1032
#define W1_LDS_ELEMS (128 * W1ROW)          // 264192 B
#define H0_LDS_ELEMS (16 * H0ROW)           //  33024 B
#define LDS_BYTES ((W1_LDS_ELEMS + H0_LDS_ELEMS) * 2)   // 297216 B < 320 KB

typedef __attribute__((ext_vector_type(16))) __bf16 v16bf;
typedef __attribute__((ext_vector_type(8)))  __bf16 v8bf;
typedef __attribute__((ext_vector_type(8)))  float  v8f;

__device__ __forceinline__ __bf16 f2bf(float f) {
  union { float f; uint32_t u; } v; v.f = f;
  uint32_t lsb = (v.u >> 16) & 1u;
  uint16_t r = (uint16_t)((v.u + 0x7FFFu + lsb) >> 16);   // round-to-nearest-even
  union { uint16_t u; __bf16 b; } o; o.u = r;
  return o.b;
}

// ---------------- fp32 -> bf16 conversion ----------------
__global__ void cvt_f32_bf16(const float* __restrict__ src, __bf16* __restrict__ dst, int n) {
  int i = blockIdx.x * blockDim.x + threadIdx.x;
  int stride = gridDim.x * blockDim.x;
  for (; i < n; i += stride) dst[i] = f2bf(src[i]);
}

__global__ void init_sync_kernel(int* s) {
  if (threadIdx.x < 8) s[threadIdx.x] = 0;
}

// ---------------- batched layer-0 GEMM: pre0 = Xbf @ W0bf^T ----------------
// X: (16384 x 1024) bf16 row-major, W: (4096 x 1024) bf16 row-major,
// P: (16384 x 4096) f32.  Each wave computes a 16(M) x 64(N) tile with 4
// independent accumulators (WMMA ILP); cross-wave occupancy hides load waits.
__global__ __launch_bounds__(TPB) void gemm0_bf16(const __bf16* __restrict__ X,
                                                  const __bf16* __restrict__ W,
                                                  float* __restrict__ P) {
  const int wave  = (blockIdx.x * TPB + threadIdx.x) >> 5;
  const int lane  = threadIdx.x & 31;
  const int mTile = wave >> 6;        // 0..1023
  const int n64   = wave & 63;        // 0..63
  const int arow  = lane & 15;
  const int hl    = lane >> 4;
  const int nLane = lane & 15;
  const int kA0 = hl * 16, kA1 = hl * 16 + 8, kB = hl * 16;
  const __bf16* xrow = X + (size_t)(mTile * 16 + arow) * INF_;

  v8f acc[4] = { v8f{}, v8f{}, v8f{}, v8f{} };
  for (int k = 0; k < INF_; k += 32) {
    // A fragment (16x32 bf16): lanes 0-15 hold K 0..15, lanes 16-31 K 16..31
    v8bf alo = *(const v8bf*)(xrow + k + kA0);
    v8bf ahi = *(const v8bf*)(xrow + k + kA1);
    const __bf16* wbase = W + (size_t)(n64 * 64 + nLane) * INF_ + k + kB;
    v16bf b0v = *(const v16bf*)(wbase + (size_t)0 * 16 * INF_);
    v16bf b1v = *(const v16bf*)(wbase + (size_t)1 * 16 * INF_);
    v16bf b2v = *(const v16bf*)(wbase + (size_t)2 * 16 * INF_);
    v16bf b3v = *(const v16bf*)(wbase + (size_t)3 * 16 * INF_);
    v16bf a;
#pragma unroll
    for (int j = 0; j < 8; ++j) { a[j] = alo[j]; a[8 + j] = ahi[j]; }
    acc[0] = __builtin_amdgcn_wmma_f32_16x16x32_bf16(false, a, false, b0v, (short)0, acc[0], false, false);
    acc[1] = __builtin_amdgcn_wmma_f32_16x16x32_bf16(false, a, false, b1v, (short)0, acc[1], false, false);
    acc[2] = __builtin_amdgcn_wmma_f32_16x16x32_bf16(false, a, false, b2v, (short)0, acc[2], false, false);
    acc[3] = __builtin_amdgcn_wmma_f32_16x16x32_bf16(false, a, false, b3v, (short)0, acc[3], false, false);
  }
#pragma unroll
  for (int s = 0; s < 4; ++s) {
    int col = n64 * 64 + s * 16 + nLane;
#pragma unroll
    for (int i = 0; i < 8; ++i) {
      int row = mTile * 16 + i + hl * 8;       // C/D layout: M = vgpr + 8*(lane>=16)
      __builtin_nontemporal_store(acc[s][i], &P[(size_t)row * FOURD + col]);
    }
  }
}

// ---------------- device-wide barrier for the persistent kernel ----------------
__device__ __forceinline__ void grid_sync(int* s, int target) {
  __threadfence();
  __syncthreads();
  if (threadIdx.x == 0) {
    int prev = __hip_atomic_fetch_add(&s[0], 1, __ATOMIC_ACQ_REL, __HIP_MEMORY_SCOPE_AGENT);
    if (prev == NBLK - 1) {
      __hip_atomic_store(&s[0], 0, __ATOMIC_RELAXED, __HIP_MEMORY_SCOPE_AGENT);
      __hip_atomic_store(&s[1], target, __ATOMIC_RELEASE, __HIP_MEMORY_SCOPE_AGENT);
    } else {
      while (__hip_atomic_load(&s[1], __ATOMIC_ACQUIRE, __HIP_MEMORY_SCOPE_AGENT) < target)
        __builtin_amdgcn_s_sleep(1);
    }
  }
  __syncthreads();
}

// ---------------- sLSTM cell (elementwise part) ----------------
__device__ __forceinline__ float cell_update(float zp, float ip, float fp, float op,
                                             float& c, float& n, float& mm) {
  float zt = tanhf(zp);
  float ot = 1.0f / (1.0f + __expf(-op));
  float mn = fmaxf(fp + mm, ip);
  float ft = __expf(fp + mm - mn);
  float it = __expf(ip - mn);
  c = ft * c + it * zt;
  n = ft * n + it;
  mm = mn;
  return ot * (c / (fabsf(n) + EPS_F));
}

// ---------------- persistent recurrence kernel ----------------
// Software-pipelined: ONE grid barrier per step. h0bf and pre1 are ping-pong
// buffers (parity t&1). Iteration t: stage h0[t&1]->LDS, GEMM(t)->pre1[t&1],
// layer-0 cell(t+1)->h0[(t+1)&1] (independent of pre1), layer-1 cell(t-1)
// from pre1[(t-1)&1], then sync. W1 slice is LDS-resident the whole time.
// GEMM uses 4 independent WMMA accumulators to break the RAW accumulate chain.
__global__ __launch_bounds__(TPB, 1) void slstm_recurrence(
    const float* __restrict__ pre0, const __bf16* __restrict__ W1bf,
    __bf16* __restrict__ h0bf, float* __restrict__ pre1,
    const float* __restrict__ h_init, const float* __restrict__ extra_init,
    const float* __restrict__ b0, const float* __restrict__ r0,
    const float* __restrict__ b1, const float* __restrict__ r1,
    float* __restrict__ out, float* __restrict__ hf, float* __restrict__ extraf,
    int* __restrict__ syncbuf) {
  extern __shared__ __align__(16) char smem[];
  __bf16* w1s  = (__bf16*)smem;                       // 128 x W1ROW
  __bf16* h0ls = (__bf16*)smem + W1_LDS_ELEMS;        // 16 x H0ROW

  const int tid   = blockIdx.x * TPB + threadIdx.x;  // 0..8191
  const int lane  = threadIdx.x & 31;
  const int wave  = tid >> 5;                        // 0..255 -> N-tile of layer-1 GEMM
  const int nBase = wave * 16;
  const int arow  = lane & 15;
  const int hl    = lane >> 4;
  const int nLane = lane & 15;
  const int kA0 = hl * 16, kA1 = hl * 16 + 8, kB = hl * 16;

  // ---- copy this block's W1 slice (128 rows) into LDS once ----
  {
    const __bf16* gw = W1bf + (size_t)blockIdx.x * 128 * DIM;
    for (int c = threadIdx.x; c < 128 * 128; c += TPB) {   // 16B chunks
      int row = c >> 7, ch = c & 127;
      v8bf v = *(const v8bf*)(gw + (size_t)row * DIM + ch * 8);
      *(v8bf*)(w1s + row * W1ROW + ch * 8) = v;
    }
  }

  // Two (b,d) elements per thread, per layer; all state in registers.
  int e[2]; e[0] = tid; e[1] = tid + NBLK * TPB;
  float rz0c[2], ri0c[2], rf0c[2], ro0c[2], bz0c[2], bi0c[2], bf0c[2], bo0c[2];
  float rz1c[2], ri1c[2], rf1c[2], ro1c[2], bz1c[2], bi1c[2], bf1c[2], bo1c[2];
  float h0s[2], c0s[2], n0s[2], m0s[2];
  float h1s[2], c1s[2], n1s[2], m1s[2];
#pragma unroll
  for (int j = 0; j < 2; ++j) {
    int d = e[j] & (DIM - 1);
    int b = e[j] >> 10;
    rz0c[j] = r0[d]; ri0c[j] = r0[DIM + d]; rf0c[j] = r0[2*DIM + d]; ro0c[j] = r0[3*DIM + d];
    bz0c[j] = b0[d]; bi0c[j] = b0[DIM + d]; bf0c[j] = b0[2*DIM + d]; bo0c[j] = b0[3*DIM + d];
    rz1c[j] = r1[d]; ri1c[j] = r1[DIM + d]; rf1c[j] = r1[2*DIM + d]; ro1c[j] = r1[3*DIM + d];
    bz1c[j] = b1[d]; bi1c[j] = b1[DIM + d]; bf1c[j] = b1[2*DIM + d]; bo1c[j] = b1[3*DIM + d];
    h0s[j] = h_init[e[j]];
    h1s[j] = h_init[BATCH * DIM + e[j]];
    size_t eb = (size_t)b * 3 * DIM + d;
    c0s[j] = extra_init[eb];             n0s[j] = extra_init[eb + DIM];
    m0s[j] = extra_init[eb + 2 * DIM];
    c1s[j] = extra_init[(size_t)BATCH * 3 * DIM + eb];
    n1s[j] = extra_init[(size_t)BATCH * 3 * DIM + eb + DIM];
    m1s[j] = extra_init[(size_t)BATCH * 3 * DIM + eb + 2 * DIM];
  }

  int bar = 0;
  // Prologue: layer-0 cell for t=0 (pre0 precomputed), publish h0(0) -> buffer 0.
#pragma unroll
  for (int j = 0; j < 2; ++j) {
    size_t base = (size_t)(e[j] >> 10) * FOURD + (e[j] & (DIM - 1));
    float zp = __builtin_nontemporal_load(&pre0[base])           + bz0c[j] + rz0c[j] * h0s[j];
    float ip = __builtin_nontemporal_load(&pre0[base + DIM])     + bi0c[j] + ri0c[j] * h0s[j];
    float fp = __builtin_nontemporal_load(&pre0[base + 2*DIM])   + bf0c[j] + rf0c[j] * h0s[j];
    float op = __builtin_nontemporal_load(&pre0[base + 3*DIM])   + bo0c[j] + ro0c[j] * h0s[j];
    h0s[j] = cell_update(zp, ip, fp, op, c0s[j], n0s[j], m0s[j]);
    h0bf[e[j]] = f2bf(h0s[j]);
  }
  ++bar; grid_sync(syncbuf, bar);

  const __bf16* arow_l = h0ls + arow * H0ROW;
  const __bf16* brow_l = w1s + ((threadIdx.x >> 5) * 16 + nLane) * W1ROW;

  for (int t = 0; t <= S_LEN; ++t) {
    const int par  = t & 1;
    const int parN = (t + 1) & 1;      // also == (t-1)&1

    if (t < S_LEN) {
      // ---- stage h0(t) into LDS (32 KB per block) ----
      const __bf16* h0cur = h0bf + par * (BATCH * DIM);
      for (int c = threadIdx.x; c < 16 * 128; c += TPB) {  // 2048 x 16B chunks
        int row = c >> 7, ch = c & 127;
        v8bf v = *(const v8bf*)(h0cur + (size_t)row * DIM + ch * 8);
        *(v8bf*)(h0ls + row * H0ROW + ch * 8) = v;
      }
      __syncthreads();

      // ---- layer-1 GEMM from LDS, 4 independent accumulator chains ----
      v8f a0 = {}, a1 = {}, a2 = {}, a3 = {};
      auto kstep = [&](int k, v8f c) -> v8f {
        v8bf alo = *(const v8bf*)(arow_l + k + kA0);
        v8bf ahi = *(const v8bf*)(arow_l + k + kA1);
        v8bf blo = *(const v8bf*)(brow_l + k + kB);
        v8bf bhi = *(const v8bf*)(brow_l + k + kB + 8);
        v16bf a, bb;
#pragma unroll
        for (int jj = 0; jj < 8; ++jj) {
          a[jj] = alo[jj];  a[8 + jj] = ahi[jj];
          bb[jj] = blo[jj]; bb[8 + jj] = bhi[jj];
        }
        return __builtin_amdgcn_wmma_f32_16x16x32_bf16(false, a, false, bb,
                                                       (short)0, c, false, false);
      };
#pragma unroll
      for (int kk = 0; kk < DIM; kk += 128) {
        a0 = kstep(kk +  0, a0);
        a1 = kstep(kk + 32, a1);
        a2 = kstep(kk + 64, a2);
        a3 = kstep(kk + 96, a3);
      }
      float* pre1cur = pre1 + par * (BATCH * FOURD);
#pragma unroll
      for (int i = 0; i < 8; ++i)
        pre1cur[(size_t)(i + hl * 8) * FOURD + nBase + nLane] =
            (a0[i] + a1[i]) + (a2[i] + a3[i]);
    }

    // ---- layer-0 cell for step t+1 (independent of pre1) -> h0 buffer (t+1)&1 ----
    if (t + 1 < S_LEN) {
      __bf16* h0nxt = h0bf + parN * (BATCH * DIM);
#pragma unroll
      for (int j = 0; j < 2; ++j) {
        size_t base0 = (size_t)(t + 1) * (BATCH * FOURD)
                     + (size_t)(e[j] >> 10) * FOURD + (e[j] & (DIM - 1));
        float zp = __builtin_nontemporal_load(&pre0[base0])         + bz0c[j] + rz0c[j] * h0s[j];
        float ip = __builtin_nontemporal_load(&pre0[base0 + DIM])   + bi0c[j] + ri0c[j] * h0s[j];
        float fp = __builtin_nontemporal_load(&pre0[base0 + 2*DIM]) + bf0c[j] + rf0c[j] * h0s[j];
        float op = __builtin_nontemporal_load(&pre0[base0 + 3*DIM]) + bo0c[j] + ro0c[j] * h0s[j];
        h0s[j] = cell_update(zp, ip, fp, op, c0s[j], n0s[j], m0s[j]);
        h0nxt[e[j]] = f2bf(h0s[j]);
      }
    }

    // ---- layer-1 cell for step t-1 (pre1[(t-1)&1] was synced last iteration) ----
    if (t >= 1) {
      const float* pre1prev = pre1 + parN * (BATCH * FOURD);
#pragma unroll
      for (int j = 0; j < 2; ++j) {
        size_t base1 = (size_t)(e[j] >> 10) * FOURD + (e[j] & (DIM - 1));
        float zp = pre1prev[base1]           + bz1c[j] + rz1c[j] * h1s[j];
        float ip = pre1prev[base1 + DIM]     + bi1c[j] + ri1c[j] * h1s[j];
        float fp = pre1prev[base1 + 2*DIM]   + bf1c[j] + rf1c[j] * h1s[j];
        float op = pre1prev[base1 + 3*DIM]   + bo1c[j] + ro1c[j] * h1s[j];
        h1s[j] = cell_update(zp, ip, fp, op, c1s[j], n1s[j], m1s[j]);
        __builtin_nontemporal_store(h1s[j], &out[(size_t)(t - 1) * (BATCH * DIM) + e[j]]);
      }
    }

    if (t < S_LEN) { ++bar; grid_sync(syncbuf, bar); }
  }

  // ---- epilogue: final h and stacked (c,n,m) states ----
#pragma unroll
  for (int j = 0; j < 2; ++j) {
    int d = e[j] & (DIM - 1); int b = e[j] >> 10;
    hf[e[j]] = h0s[j];
    hf[BATCH * DIM + e[j]] = h1s[j];
    size_t eb = (size_t)b * 3 * DIM + d;
    extraf[eb] = c0s[j]; extraf[eb + DIM] = n0s[j]; extraf[eb + 2*DIM] = m0s[j];
    size_t eb1 = (size_t)BATCH * 3 * DIM + eb;
    extraf[eb1] = c1s[j]; extraf[eb1 + DIM] = n1s[j]; extraf[eb1 + 2*DIM] = m1s[j];
  }
}

// ---------------- host launcher ----------------
extern "C" void kernel_launch(void* const* d_in, const int* in_sizes, int n_in,
                              void* d_out, int out_size, void* d_ws, size_t ws_size,
                              hipStream_t stream) {
  (void)in_sizes; (void)n_in; (void)out_size; (void)ws_size;
  const float* input  = (const float*)d_in[0];   // (1024,16,1024)
  const float* h0     = (const float*)d_in[1];   // (2,16,1024)
  const float* extra0 = (const float*)d_in[2];   // (2,16,3072)
  const float* W0     = (const float*)d_in[3];   // (4096,1024)
  const float* b0     = (const float*)d_in[4];
  const float* r0     = (const float*)d_in[5];
  const float* W1     = (const float*)d_in[6];
  const float* b1     = (const float*)d_in[7];
  const float* r1     = (const float*)d_in[8];

  float* out    = (float*)d_out;                 // (1024,16,1024)
  float* hf     = out + (size_t)S_LEN * BATCH * DIM;       // (2,16,1024)
  float* extraf = hf + 2 * BATCH * DIM;                    // (2,16,3072)

  // workspace carve (aligned 256B). Total ~320 MB.
  uintptr_t p = (uintptr_t)d_ws;
  auto take = [&](size_t bytes) -> void* {
    uintptr_t cur = (p + 255) & ~(uintptr_t)255;
    p = cur + bytes;
    return (void*)cur;
  };
  __bf16* xbf   = (__bf16*)take((size_t)S_LEN * BATCH * INF_ * 2);   // 32 MB
  __bf16* w0bf  = (__bf16*)take((size_t)FOURD * INF_ * 2);           //  8 MB
  __bf16* w1bf  = (__bf16*)take((size_t)FOURD * DIM * 2);            //  8 MB
  __bf16* h0bfp = (__bf16*)take((size_t)2 * BATCH * DIM * 2);        // 64 KB (ping-pong)
  float*  pre1  = (float*)take((size_t)2 * BATCH * FOURD * 4);       // 512 KB (ping-pong)
  int*    syncb = (int*)take(256);
  float*  pre0  = (float*)take((size_t)S_LEN * BATCH * FOURD * 4);   // 256 MB

  cvt_f32_bf16<<<4096, TPB, 0, stream>>>(input, xbf, S_LEN * BATCH * INF_);
  cvt_f32_bf16<<<1024, TPB, 0, stream>>>(W0, w0bf, FOURD * INF_);
  cvt_f32_bf16<<<1024, TPB, 0, stream>>>(W1, w1bf, FOURD * DIM);
  init_sync_kernel<<<1, 32, 0, stream>>>(syncb);

  // Batched layer-0 GEMM: 1024 M-tiles x 64 N64-tiles = 65536 waves = 8192 blocks
  gemm0_bf16<<<8192, TPB, 0, stream>>>(xbf, w0bf, pre0);

  // Persistent recurrence: 32 blocks x 256 threads, 290 KB dynamic LDS each
  slstm_recurrence<<<NBLK, TPB, LDS_BYTES, stream>>>(pre0, w1bf, h0bfp, pre1,
                                                     h0, extra0, b0, r0, b1, r1,
                                                     out, hf, extraf, syncb);
}
